// AttentionAggregator_61512521613321
// MI455X (gfx1250) — matrix operand order repaired and checked by reading, not compiled, not run on hardware
//
#include <hip/hip_runtime.h>

#define BATCH 8192
#define NTOK  16
#define DMODEL 128
#define NHEAD 4
#define HD    32
#define HID   256

typedef __attribute__((ext_vector_type(16))) __bf16 v16bf;
typedef __attribute__((ext_vector_type(8)))  float  v8f;

__device__ __forceinline__ unsigned short f2bf(float f) {
    unsigned int u = __float_as_uint(f);
    u += 0x7FFFu + ((u >> 16) & 1u);           // round-to-nearest-even
    return (unsigned short)(u >> 16);
}

// ---- WMMA fragment loaders (contiguous k = 16*(l>>4)+s per lane, wave32) -------------------

// A (16x32 bf16), row-major source, stride ld (elems). lane: m=l&15, K-base=16*(l>>4)
__device__ __forceinline__ v16bf load_a(const unsigned short* p, int ld, int k0) {
    const int l  = threadIdx.x & 31;
    const int m  = l & 15;
    const int kb = (l >> 4) * 16;
    union { v16bf v; unsigned int u[8]; } f;
    const unsigned int* p0 = (const unsigned int*)(p + m * ld + k0 + kb);
#pragma unroll
    for (int r = 0; r < 8; ++r) { f.u[r] = p0[r]; }
    return f.v;
}

// B (32x16 bf16) from row-major weight W[n][k] (out = act @ W^T): per-lane 16
// consecutive K values -> two b128 loads. lane: n=l&15, K-base=16*(l>>4)
__device__ __forceinline__ v16bf load_b_w(const unsigned short* W, int ldw, int n0, int k0) {
    const int l  = threadIdx.x & 31;
    const int n  = l & 15;
    const int kb = (l >> 4) * 16;
    union { v16bf v; uint4 q[2]; } f;
    const uint4* p = (const uint4*)(W + (size_t)(n0 + n) * ldw + k0 + kb);
    f.q[0] = p[0]; f.q[1] = p[1];
    return f.v;
}

// B (32x16) where B[k][n] = p[n*ld + k0 + k] (e.g. K^T for q.k^T): contiguous per lane
__device__ __forceinline__ v16bf load_b_ct(const unsigned short* p, int ld, int k0) {
    const int l  = threadIdx.x & 31;
    const int n  = l & 15;
    const int kb = (l >> 4) * 16;
    union { v16bf v; unsigned int u[8]; } f;
    const unsigned int* q = (const unsigned int*)(p + n * ld + k0 + kb);
#pragma unroll
    for (int r = 0; r < 8; ++r) f.u[r] = q[r];
    return f.v;
}

// B (32x16) where B[k][n] = p[k*ld + d0 + n] (v for attn@v): strided gather from LDS
__device__ __forceinline__ v16bf load_b_st(const unsigned short* p, int ld, int d0) {
    const int l  = threadIdx.x & 31;
    const int n  = l & 15;
    const int kb = (l >> 4) * 16;
    union { v16bf v; unsigned int u[8]; } f;
#pragma unroll
    for (int r = 0; r < 8; ++r) {
        unsigned int lo = p[(kb + 2 * r) * ld + d0 + n];
        unsigned int hi = p[(kb + 2 * r + 1) * ld + d0 + n];
        f.u[r] = lo | (hi << 16);
    }
    return f.v;
}

// C/D f32: lane n=l&15, vgpr r -> m = r + 8*(l>>4). Store bf16 row-major to LDS.
__device__ __forceinline__ void store_c(unsigned short* p, int ld, int n0, v8f c, bool relu) {
    const int l  = threadIdx.x & 31;
    const int n  = l & 15;
    const int mb = (l >> 4) * 8;
#pragma unroll
    for (int r = 0; r < 8; ++r) {
        float x = c[r];
        if (relu) x = fmaxf(x, 0.f);
        p[(mb + r) * ld + n0 + n] = f2bf(x);
    }
}

__device__ __forceinline__ v8f wmma_bf16(v16bf a, v16bf b, v8f c) {
    return __builtin_amdgcn_wmma_f32_16x16x32_bf16(false, a, false, b, (short)0, c, false, false);
}

// 16 x 16 output tile of act(16xK) @ W^T, K-loop over 32-wide steps
__device__ __forceinline__ v8f gemm_tile(const unsigned short* act, int lda,
                                         const unsigned short* W, int ldw,
                                         int n0, int K) {
    v8f c = {0.f, 0.f, 0.f, 0.f, 0.f, 0.f, 0.f, 0.f};
#pragma unroll
    for (int k0 = 0; k0 < K; k0 += 32) {
        v16bf a = load_a(act, lda, k0);
        v16bf b = load_b_w(W, ldw, n0, k0);
        c = wmma_bf16(a, b, c);
    }
    return c;
}

__device__ __forceinline__ v8f add_bias(v8f c, const float* bias, int n0) {
    float bv = bias[n0 + (threadIdx.x & 15)];
#pragma unroll
    for (int r = 0; r < 8; ++r) c[r] += bv;
    return c;
}

// ---- weight f32 -> bf16 prep ----------------------------------------------
__global__ void cvt_bf16(const float* __restrict__ src, unsigned short* __restrict__ dst, int n) {
    int i = blockIdx.x * 256 + threadIdx.x;
    if (i < n) dst[i] = f2bf(src[i]);
}

// ---- fused attention aggregator: one block (4 waves) per batch item -------
__launch_bounds__(128)
__global__ void aa_main(const float* __restrict__ obs, const float* __restrict__ msgs,
                        const float* __restrict__ qp_b, const float* __restrict__ in_b,
                        const float* __restrict__ op_b, const float* __restrict__ mlp_b1,
                        const float* __restrict__ mlp_b2,
                        const unsigned short* __restrict__ w_qp,
                        const unsigned short* __restrict__ w_in,
                        const unsigned short* __restrict__ w_op,
                        const unsigned short* __restrict__ w1,
                        const unsigned short* __restrict__ w2,
                        float* __restrict__ out) {
    __shared__ unsigned short obs_s[NTOK * DMODEL];
    __shared__ unsigned short msg_s[NTOK * DMODEL];
    __shared__ unsigned short qb_s [NTOK * DMODEL];
    __shared__ unsigned short q_s  [NTOK * DMODEL];
    __shared__ unsigned short k_s  [NTOK * DMODEL];
    __shared__ unsigned short v_s  [32   * DMODEL];   // rows 16..31 zero (K-pad)
    __shared__ unsigned short at_s [NHEAD * NTOK * 32]; // attn, cols 16..31 zero
    __shared__ unsigned short cx_s [NTOK * DMODEL];
    __shared__ unsigned short ao_s [NTOK * DMODEL];
    __shared__ unsigned short h_s  [NTOK * HID];

    const int b    = blockIdx.x;
    const int tid  = threadIdx.x;
    const int w    = tid >> 5;
    const int lane = tid & 31;

    // Stage 0: stream obs/msgs (HBM, once) -> bf16 LDS; zero pad regions
    const float* ob = obs  + (size_t)b * NTOK * DMODEL;
    const float* mg = msgs + (size_t)b * NTOK * DMODEL;
    unsigned int* os32 = (unsigned int*)obs_s;
    unsigned int* ms32 = (unsigned int*)msg_s;
#pragma unroll
    for (int i = tid * 4; i < NTOK * DMODEL; i += 128 * 4) {
        float4 o4 = *(const float4*)(ob + i);
        float4 m4 = *(const float4*)(mg + i);
        os32[i / 2]     = (unsigned int)f2bf(o4.x) | ((unsigned int)f2bf(o4.y) << 16);
        os32[i / 2 + 1] = (unsigned int)f2bf(o4.z) | ((unsigned int)f2bf(o4.w) << 16);
        ms32[i / 2]     = (unsigned int)f2bf(m4.x) | ((unsigned int)f2bf(m4.y) << 16);
        ms32[i / 2 + 1] = (unsigned int)f2bf(m4.z) | ((unsigned int)f2bf(m4.w) << 16);
    }
    {
        unsigned int* vz = (unsigned int*)(v_s + 16 * DMODEL);
        unsigned int* az = (unsigned int*)at_s;
        for (int i = tid; i < 1024; i += 128) { vz[i] = 0u; az[i] = 0u; }
    }
    __syncthreads();

    // Stage 1: q_base = obs@qp_w^T + qp_b ; k,v = msgs@{wk,wv}^T + {bk,bv}
#pragma unroll
    for (int t = 0; t < 2; ++t) {
        const int n0 = (2 * w + t) * 16;
        v8f c = gemm_tile(obs_s, DMODEL, w_qp, DMODEL, n0, DMODEL);
        store_c(qb_s, DMODEL, n0, add_bias(c, qp_b, n0), false);

        c = gemm_tile(msg_s, DMODEL, w_in + DMODEL * DMODEL, DMODEL, n0, DMODEL);
        store_c(k_s, DMODEL, n0, add_bias(c, in_b + DMODEL, n0), false);

        c = gemm_tile(msg_s, DMODEL, w_in + 2 * DMODEL * DMODEL, DMODEL, n0, DMODEL);
        store_c(v_s, DMODEL, n0, add_bias(c, in_b + 2 * DMODEL, n0), false);
    }
    __syncthreads();

    // Stage 2: q = q_base @ wq^T + bq
#pragma unroll
    for (int t = 0; t < 2; ++t) {
        const int n0 = (2 * w + t) * 16;
        v8f c = gemm_tile(qb_s, DMODEL, w_in, DMODEL, n0, DMODEL);
        store_c(q_s, DMODEL, n0, add_bias(c, in_b, n0), false);
    }
    __syncthreads();

    // Stage 3: per-head attention (head h = wave id). scores = q.k^T / sqrt(HD)
    {
        const int h = w;
        v16bf aq = load_a(q_s, DMODEL, h * HD);
        v16bf bk = load_b_ct(k_s, DMODEL, h * HD);
        v8f c = {0.f, 0.f, 0.f, 0.f, 0.f, 0.f, 0.f, 0.f};
        c = wmma_bf16(aq, bk, c);
        const float scale = 0.17677669529663687f;  // 1/sqrt(32)
        const int n  = lane & 15;
        const int mb = (lane >> 4) * 8;
#pragma unroll
        for (int r = 0; r < 8; ++r) {
            float x = c[r] * scale;
            if (n == mb + r) x = -1e30f;           // self mask (diagonal)
            float mx = x;
#pragma unroll
            for (int off = 1; off < 16; off <<= 1) mx = fmaxf(mx, __shfl_xor(mx, off, 32));
            float e = __expf(x - mx);
            float sum = e;
#pragma unroll
            for (int off = 1; off < 16; off <<= 1) sum += __shfl_xor(sum, off, 32);
            at_s[h * NTOK * 32 + (mb + r) * 32 + n] = f2bf(e / sum);
        }
    }
    __syncthreads();

    // Stage 4: ctx = attn @ v  (A K-padded to 32 with zeros; v rows 16..31 zero)
    {
        const int h = w;
        v16bf aa = load_a(at_s + h * NTOK * 32, 32, 0);
#pragma unroll
        for (int dt = 0; dt < 2; ++dt) {
            const int d0 = h * HD + dt * 16;
            v16bf bv = load_b_st(v_s, DMODEL, d0);
            v8f c = {0.f, 0.f, 0.f, 0.f, 0.f, 0.f, 0.f, 0.f};
            c = wmma_bf16(aa, bv, c);
            store_c(cx_s, DMODEL, d0, c, false);
        }
    }
    __syncthreads();

    // Stage 5: attn_out = ctx @ op_w^T + op_b
#pragma unroll
    for (int t = 0; t < 2; ++t) {
        const int n0 = (2 * w + t) * 16;
        v8f c = gemm_tile(cx_s, DMODEL, w_op, DMODEL, n0, DMODEL);
        store_c(ao_s, DMODEL, n0, add_bias(c, op_b, n0), false);
    }
    __syncthreads();

    // Stage 6: h = relu(attn_out @ w1^T + b1)   (16 x 256)
#pragma unroll
    for (int t = 0; t < 4; ++t) {
        const int n0 = (w * 4 + t) * 16;
        v8f c = gemm_tile(ao_s, DMODEL, w1, DMODEL, n0, DMODEL);
        store_c(h_s, HID, n0, add_bias(c, mlp_b1, n0), true);
    }
    __syncthreads();

    // Stage 7: out = h @ w2^T + b2  -> f32 HBM (written once)
    float* op = out + (size_t)b * NTOK * DMODEL;
#pragma unroll
    for (int t = 0; t < 2; ++t) {
        const int n0 = (2 * w + t) * 16;
        v8f c = gemm_tile(h_s, HID, w2, HID, n0, HID);
        c = add_bias(c, mlp_b2, n0);
        const int n  = lane & 15;
        const int mb = (lane >> 4) * 8;
#pragma unroll
        for (int r = 0; r < 8; ++r) op[(mb + r) * DMODEL + n0 + n] = c[r];
    }
}

extern "C" void kernel_launch(void* const* d_in, const int* in_sizes, int n_in,
                              void* d_out, int out_size, void* d_ws, size_t ws_size,
                              hipStream_t stream) {
    const float* obs   = (const float*)d_in[0];
    const float* msgs  = (const float*)d_in[1];
    const float* qp_w  = (const float*)d_in[2];
    const float* qp_b  = (const float*)d_in[3];
    const float* in_w  = (const float*)d_in[4];
    const float* in_b  = (const float*)d_in[5];
    const float* op_w  = (const float*)d_in[6];
    const float* op_b  = (const float*)d_in[7];
    const float* w1    = (const float*)d_in[8];
    const float* b1    = (const float*)d_in[9];
    const float* w2    = (const float*)d_in[10];
    const float* b2    = (const float*)d_in[11];

    unsigned short* ws = (unsigned short*)d_ws;
    unsigned short* qpw_bf = ws;              // 128*128       = 16384
    unsigned short* inw_bf = ws + 16384;      // 384*128       = 49152
    unsigned short* opw_bf = ws + 65536;      // 128*128       = 16384
    unsigned short* w1_bf  = ws + 81920;      // 256*128       = 32768
    unsigned short* w2_bf  = ws + 114688;     // 128*256       = 32768

    cvt_bf16<<<(16384 + 255) / 256, 256, 0, stream>>>(qp_w, qpw_bf, 16384);
    cvt_bf16<<<(49152 + 255) / 256, 256, 0, stream>>>(in_w, inw_bf, 49152);
    cvt_bf16<<<(16384 + 255) / 256, 256, 0, stream>>>(op_w, opw_bf, 16384);
    cvt_bf16<<<(32768 + 255) / 256, 256, 0, stream>>>(w1, w1_bf, 32768);
    cvt_bf16<<<(32768 + 255) / 256, 256, 0, stream>>>(w2, w2_bf, 32768);

    aa_main<<<BATCH, 128, 0, stream>>>(obs, msgs, qp_b, in_b, op_b, b1, b2,
                                       qpw_bf, inw_bf, opw_bf, w1_bf, w2_bf,
                                       (float*)d_out);
}
